// Transformer_with_ib_89386859364792
// MI455X (gfx1250) — matrix operand order repaired and checked by reading, not compiled, not run on hardware
//
#include <hip/hip_runtime.h>
#include <hip/hip_bf16.h>
#include <math.h>

#define LAYERS 6
#define DMODEL 1024
#define HIDN   4096
#define HEADS  16
#define BATCH  2
#define SEQ    2048
#define HD     64              // DMODEL / HEADS
#define MROWS  (BATCH * SEQ)   // 4096
#define D3     (3 * DMODEL)

typedef __attribute__((ext_vector_type(16))) __bf16         bf16x16;
typedef __attribute__((ext_vector_type(8)))  float          f32x8;
typedef __attribute__((ext_vector_type(8)))  unsigned short u16x8;

union FragBF {
    bf16x16        v;
    u16x8          u[2];
    unsigned short s[16];
};

__device__ __forceinline__ unsigned short f2bf(float f) {
    unsigned int u = __float_as_uint(f);
    u += 0x7FFFu + ((u >> 16) & 1u);      // round-to-nearest-even
    return (unsigned short)(u >> 16);
}

// ---------------------------------------------------------------------------
// KL divergence: -0.5 * sum(1 + logD - mu^2 - exp(logD)) over both IB gates.
// ---------------------------------------------------------------------------
__global__ void __launch_bounds__(256) kld_kernel(
        const float* __restrict__ mu1, const float* __restrict__ ld1,
        const float* __restrict__ mu2, const float* __restrict__ ld2,
        float* __restrict__ out) {
    __shared__ float red[256];
    float s = 0.f;
    for (int i = threadIdx.x; i < LAYERS * DMODEL; i += 256) {
        float m = mu1[i], d = ld1[i];
        s += 1.f + d - m * m - expf(d);
        m = mu2[i]; d = ld2[i];
        s += 1.f + d - m * m - expf(d);
    }
    red[threadIdx.x] = s;
    __syncthreads();
    for (int off = 128; off > 0; off >>= 1) {
        if (threadIdx.x < off) red[threadIdx.x] += red[threadIdx.x + off];
        __syncthreads();
    }
    if (threadIdx.x == 0) out[0] = -0.5f * red[0];
}

// ---------------------------------------------------------------------------
// h = layernorm(x) * z  (z = hard VIB gate), stored bf16. One row per block.
// ---------------------------------------------------------------------------
__global__ void __launch_bounds__(256) ln_ib_kernel(
        const float* __restrict__ x,
        const float* __restrict__ w, const float* __restrict__ b,
        const float* __restrict__ mu, const float* __restrict__ logD,
        unsigned short* __restrict__ out) {
    __shared__ float red[256];
    const int row = blockIdx.x;
    const int tid = threadIdx.x;
    const float* xr = x + (size_t)row * DMODEL;

    float v[4];
    float s = 0.f;
#pragma unroll
    for (int j = 0; j < 4; ++j) { v[j] = xr[tid + j * 256]; s += v[j]; }
    red[tid] = s;
    __syncthreads();
    for (int off = 128; off > 0; off >>= 1) {
        if (tid < off) red[tid] += red[tid + off];
        __syncthreads();
    }
    const float mean = red[0] * (1.f / DMODEL);
    __syncthreads();

    s = 0.f;
#pragma unroll
    for (int j = 0; j < 4; ++j) { float c = v[j] - mean; s += c * c; }
    red[tid] = s;
    __syncthreads();
    for (int off = 128; off > 0; off >>= 1) {
        if (tid < off) red[tid] += red[tid + off];
        __syncthreads();
    }
    const float rstd = rsqrtf(red[0] * (1.f / DMODEL) + 1e-5f);

#pragma unroll
    for (int j = 0; j < 4; ++j) {
        const int d = tid + j * 256;
        const float mv = mu[d];
        const float la = logD[d] - logf(mv * mv + 1e-8f);
        const float z  = (la < 0.f) ? mv : 0.f;
        const float h  = ((v[j] - mean) * rstd * w[d] + b[d]) * z;
        out[(size_t)row * DMODEL + d] = f2bf(h);
    }
}

// ---------------------------------------------------------------------------
// fp32 -> bf16 weight convert (n multiple of 4).
// ---------------------------------------------------------------------------
__global__ void __launch_bounds__(256) cvt_kernel(
        const float* __restrict__ in, unsigned short* __restrict__ out, int n) {
    const int i = (blockIdx.x * 256 + threadIdx.x) * 4;
    if (i < n) {
        const float4 f = *reinterpret_cast<const float4*>(in + i);
        ushort4 o;
        o.x = f2bf(f.x); o.y = f2bf(f.y); o.z = f2bf(f.z); o.w = f2bf(f.w);
        *reinterpret_cast<ushort4*>(out + i) = o;
    }
}

// ---------------------------------------------------------------------------
// C[M,Nn] = A[M,K] @ W[Nn,K]^T  via V_WMMA_F32_16X16X32_BF16.
// Block = 256 threads (8 waves), tile 128x128; wave tile 64(M) x 32(N).
// MODE 0: store bf16.  MODE 1: +bias +residual, store fp32.  MODE 2: +bias, GELU, store bf16.
// ---------------------------------------------------------------------------
template <int MODE>
__global__ void __launch_bounds__(256) gemm_bf16_kernel(
        const unsigned short* __restrict__ A,
        const unsigned short* __restrict__ W,
        const float* __restrict__ bias,
        const float* __restrict__ resid,
        unsigned short* __restrict__ outb,
        float* __restrict__ outf,
        int Nn, int K) {
    const int lane = threadIdx.x & 31;
    const int wave = threadIdx.x >> 5;
    const int wm   = wave >> 2;                       // 0..1
    const int wn   = wave & 3;                        // 0..3
    const int m_wave = blockIdx.y * 128 + wm * 64;
    const int n_wave = blockIdx.x * 128 + wn * 32;
    const int half = lane >> 4;
    const int l15  = lane & 15;

    // per-lane row base pointers (invariant over k)
    const unsigned short* arow[4];
#pragma unroll
    for (int im = 0; im < 4; ++im)
        arow[im] = A + (size_t)(m_wave + im * 16 + l15) * K + half * 16;
    const unsigned short* wrow[2];
#pragma unroll
    for (int in2 = 0; in2 < 2; ++in2)
        wrow[in2] = W + (size_t)(n_wave + in2 * 16 + l15) * K + half * 16;

    f32x8 acc[4][2];
#pragma unroll
    for (int i = 0; i < 4; ++i)
#pragma unroll
        for (int j = 0; j < 2; ++j)
#pragma unroll
            for (int r = 0; r < 8; ++r) acc[i][j][r] = 0.f;

    for (int k0 = 0; k0 < K; k0 += 32) {
        FragBF a[4];
#pragma unroll
        for (int im = 0; im < 4; ++im) {
            const unsigned short* p = arow[im] + k0;
            a[im].u[0] = *reinterpret_cast<const u16x8*>(p);
            a[im].u[1] = *reinterpret_cast<const u16x8*>(p + 8);
        }
        FragBF bf[2];
#pragma unroll
        for (int in2 = 0; in2 < 2; ++in2) {
            const unsigned short* p = wrow[in2] + k0;
            bf[in2].u[0] = *reinterpret_cast<const u16x8*>(p);
            bf[in2].u[1] = *reinterpret_cast<const u16x8*>(p + 8);
        }
        // warm WGP$/L2 for the next k-chunk (global_prefetch_b8)
        if (k0 + 32 < K) {
#pragma unroll
            for (int im = 0; im < 4; ++im)
                __builtin_prefetch(arow[im] + k0 + 32, 0, 3);
#pragma unroll
            for (int in2 = 0; in2 < 2; ++in2)
                __builtin_prefetch(wrow[in2] + k0 + 32, 0, 3);
        }
#pragma unroll
        for (int im = 0; im < 4; ++im)
#pragma unroll
            for (int in2 = 0; in2 < 2; ++in2)
                acc[im][in2] = __builtin_amdgcn_wmma_f32_16x16x32_bf16(
                    false, a[im].v, false, bf[in2].v, (short)0, acc[im][in2], false, false);
    }

#pragma unroll
    for (int im = 0; im < 4; ++im)
#pragma unroll
        for (int in2 = 0; in2 < 2; ++in2) {
            const int col = n_wave + in2 * 16 + l15;
#pragma unroll
            for (int r = 0; r < 8; ++r) {
                const int row = m_wave + im * 16 + r + half * 8;
                float vv = acc[im][in2][r];
                if constexpr (MODE == 0) {
                    outb[(size_t)row * Nn + col] = f2bf(vv);
                } else if constexpr (MODE == 1) {
                    vv += bias[col] + resid[(size_t)row * Nn + col];
                    outf[(size_t)row * Nn + col] = vv;
                } else {
                    vv += bias[col];
                    const float g = 0.5f * vv * (1.f + erff(vv * 0.70710678118654752f));
                    outb[(size_t)row * Nn + col] = f2bf(g);
                }
            }
        }
}

// ---------------------------------------------------------------------------
// Flash attention: one wave per (batch, head, 16-query tile).
// S = Q K^T * D^-0.5, online softmax over 32-key blocks, O += P V.
// qkv layout: [M, 3*D], q at +0, k at +DMODEL, v at +2*DMODEL, head offset h*HD.
// launch_bounds(32): one wave per block -> full VGPR budget, no spills.
// ---------------------------------------------------------------------------
__global__ void __launch_bounds__(32) attn_kernel(
        const unsigned short* __restrict__ qkv,
        unsigned short* __restrict__ outp) {
    __shared__ __align__(16) unsigned short lds_p[16 * 32];
    const int lane = threadIdx.x;      // 32-thread block == one wave
    const int half = lane >> 4;
    const int l15  = lane & 15;
    const int nqt  = SEQ / 16;         // 128
    const int qt = blockIdx.x % nqt;
    const int h  = (blockIdx.x / nqt) % HEADS;
    const int bi = blockIdx.x / (nqt * HEADS);
    const int m0 = qt * 16;

    // Q fragments (16 x 64 -> two 16x32 A-frags)
    FragBF q[2];
#pragma unroll
    for (int c = 0; c < 2; ++c) {
        const unsigned short* p =
            qkv + (size_t)(bi * SEQ + m0 + l15) * D3 + h * HD + c * 32 + half * 16;
        q[c].u[0] = *reinterpret_cast<const u16x8*>(p);
        q[c].u[1] = *reinterpret_cast<const u16x8*>(p + 8);
    }

    // loop-invariant base pointers
    const unsigned short* kbase =
        qkv + (size_t)(bi * SEQ + l15) * D3 + DMODEL + h * HD + half * 16;
    const unsigned short* vbase =
        qkv + (size_t)(bi * SEQ + half * 16) * D3 + 2 * DMODEL + h * HD + l15;

    f32x8 o[4];
#pragma unroll
    for (int dd = 0; dd < 4; ++dd)
#pragma unroll
        for (int r = 0; r < 8; ++r) o[dd][r] = 0.f;

    float mrun[8], lrun[8];
#pragma unroll
    for (int r = 0; r < 8; ++r) { mrun[r] = -1e30f; lrun[r] = 0.f; }

    for (int kb = 0; kb < SEQ; kb += 32) {
        f32x8 s[2];
#pragma unroll
        for (int t = 0; t < 2; ++t) {
            f32x8 c0;
#pragma unroll
            for (int r = 0; r < 8; ++r) c0[r] = 0.f;
#pragma unroll
            for (int c = 0; c < 2; ++c) {
                FragBF kf;
                const unsigned short* p = kbase + (size_t)(kb + t * 16) * D3 + c * 32;
                kf.u[0] = *reinterpret_cast<const u16x8*>(p);
                kf.u[1] = *reinterpret_cast<const u16x8*>(p + 8);
                c0 = __builtin_amdgcn_wmma_f32_16x16x32_bf16(
                    false, q[c].v, false, kf.v, (short)0, c0, false, false);
            }
#pragma unroll
            for (int r = 0; r < 8; ++r) c0[r] *= 0.03125f;   // D^-0.5 = 1/32
            s[t] = c0;
        }

        // online softmax (rows split by lane-half, matching C-layout)
#pragma unroll
        for (int r = 0; r < 8; ++r) {
            float mx = fmaxf(s[0][r], s[1][r]);
#pragma unroll
            for (int off = 1; off < 16; off <<= 1) mx = fmaxf(mx, __shfl_xor(mx, off, 32));
            const float mnew = fmaxf(mrun[r], mx);
            const float alpha = expf(mrun[r] - mnew);
            const float p0 = expf(s[0][r] - mnew);
            const float p1 = expf(s[1][r] - mnew);
            s[0][r] = p0; s[1][r] = p1;
            float sum = p0 + p1;
#pragma unroll
            for (int off = 1; off < 16; off <<= 1) sum += __shfl_xor(sum, off, 32);
            lrun[r] = lrun[r] * alpha + sum;
            mrun[r] = mnew;
#pragma unroll
            for (int dd = 0; dd < 4; ++dd) o[dd][r] *= alpha;
        }

        // P: C-layout -> row-major 16x32 bf16 in LDS
#pragma unroll
        for (int t = 0; t < 2; ++t)
#pragma unroll
            for (int r = 0; r < 8; ++r)
                lds_p[(r + half * 8) * 32 + t * 16 + l15] = f2bf(s[t][r]);
        __syncthreads();

        // reload P in A-fragment layout
        FragBF pf;
        {
            const unsigned short* p = &lds_p[l15 * 32 + half * 16];
            pf.u[0] = *reinterpret_cast<const u16x8*>(p);
            pf.u[1] = *reinterpret_cast<const u16x8*>(p + 8);
        }
        __syncthreads();

        // O += P @ V  (V as 32x16 B-frags per 16-wide d chunk)
        const unsigned short* vkb = vbase + (size_t)kb * D3;
#pragma unroll
        for (int dd = 0; dd < 4; ++dd) {
            FragBF vf;
            const unsigned short* vp = vkb + dd * 16;
#pragma unroll
            for (int j = 0; j < 16; ++j)
                vf.s[j] = vp[(size_t)j * D3];
            o[dd] = __builtin_amdgcn_wmma_f32_16x16x32_bf16(
                false, pf.v, false, vf.v, (short)0, o[dd], false, false);
        }
    }

#pragma unroll
    for (int r = 0; r < 8; ++r) {
        const float inv = 1.f / lrun[r];
        const int rowg = bi * SEQ + m0 + r + half * 8;
#pragma unroll
        for (int dd = 0; dd < 4; ++dd)
            outp[(size_t)rowg * DMODEL + h * HD + dd * 16 + l15] = f2bf(o[dd][r] * inv);
    }
}

// ---------------------------------------------------------------------------
extern "C" void kernel_launch(void* const* d_in, const int* in_sizes, int n_in,
                              void* d_out, int out_size, void* d_ws, size_t ws_size,
                              hipStream_t stream) {
    (void)in_sizes; (void)n_in; (void)out_size; (void)ws_size;
    const float* x_in = (const float*)d_in[0];
    const float* ln1w = (const float*)d_in[1];
    const float* ln1b = (const float*)d_in[2];
    const float* mu1  = (const float*)d_in[3];
    const float* ld1  = (const float*)d_in[4];
    const float* qkvw = (const float*)d_in[5];
    const float* outw = (const float*)d_in[6];
    const float* outb = (const float*)d_in[7];
    const float* ln2w = (const float*)d_in[8];
    const float* ln2b = (const float*)d_in[9];
    const float* mu2  = (const float*)d_in[10];
    const float* ld2  = (const float*)d_in[11];
    const float* ffw1 = (const float*)d_in[12];
    const float* ffb1 = (const float*)d_in[13];
    const float* ffw2 = (const float*)d_in[14];
    const float* ffb2 = (const float*)d_in[15];

    char* ws = (char*)d_ws;
    size_t off = 0;
    float* x_ws = (float*)(ws + off);                 off += (size_t)MROWS * DMODEL * 4;
    unsigned short* h_bf   = (unsigned short*)(ws + off); off += (size_t)MROWS * DMODEL * 2;
    unsigned short* qkv_bf = (unsigned short*)(ws + off); off += (size_t)MROWS * D3 * 2;
    unsigned short* ao_bf  = (unsigned short*)(ws + off); off += (size_t)MROWS * DMODEL * 2;
    unsigned short* ff_bf  = (unsigned short*)(ws + off); off += (size_t)MROWS * HIDN * 2;
    unsigned short* w_bf   = (unsigned short*)(ws + off); off += (size_t)HIDN * DMODEL * 2;

    hipMemcpyAsync(x_ws, x_in, (size_t)MROWS * DMODEL * 4, hipMemcpyDeviceToDevice, stream);
    kld_kernel<<<1, 256, 0, stream>>>(mu1, ld1, mu2, ld2, (float*)d_out + (size_t)MROWS * DMODEL);

    for (int l = 0; l < LAYERS; ++l) {
        // --- attention block ---
        ln_ib_kernel<<<MROWS, 256, 0, stream>>>(x_ws, ln1w + l * DMODEL, ln1b + l * DMODEL,
                                                mu1 + l * DMODEL, ld1 + l * DMODEL, h_bf);
        const int nq = D3 * DMODEL;
        cvt_kernel<<<(nq / 4 + 255) / 256, 256, 0, stream>>>(qkvw + (size_t)l * nq, w_bf, nq);
        gemm_bf16_kernel<0><<<dim3(D3 / 128, MROWS / 128), 256, 0, stream>>>(
            h_bf, w_bf, nullptr, nullptr, qkv_bf, nullptr, D3, DMODEL);
        attn_kernel<<<BATCH * HEADS * (SEQ / 16), 32, 0, stream>>>(qkv_bf, ao_bf);
        const int no = DMODEL * DMODEL;
        cvt_kernel<<<(no / 4 + 255) / 256, 256, 0, stream>>>(outw + (size_t)l * no, w_bf, no);
        gemm_bf16_kernel<1><<<dim3(DMODEL / 128, MROWS / 128), 256, 0, stream>>>(
            ao_bf, w_bf, outb + l * DMODEL, x_ws, nullptr, x_ws, DMODEL, DMODEL);

        // --- feedforward block ---
        ln_ib_kernel<<<MROWS, 256, 0, stream>>>(x_ws, ln2w + l * DMODEL, ln2b + l * DMODEL,
                                                mu2 + l * DMODEL, ld2 + l * DMODEL, h_bf);
        const int n1 = HIDN * DMODEL;
        cvt_kernel<<<(n1 / 4 + 255) / 256, 256, 0, stream>>>(ffw1 + (size_t)l * n1, w_bf, n1);
        gemm_bf16_kernel<2><<<dim3(HIDN / 128, MROWS / 128), 256, 0, stream>>>(
            h_bf, w_bf, ffb1 + l * HIDN, nullptr, ff_bf, nullptr, HIDN, DMODEL);
        cvt_kernel<<<(n1 / 4 + 255) / 256, 256, 0, stream>>>(ffw2 + (size_t)l * n1, w_bf, n1);
        gemm_bf16_kernel<1><<<dim3(DMODEL / 128, MROWS / 128), 256, 0, stream>>>(
            ff_bf, w_bf, ffb2 + l * DMODEL, x_ws, nullptr, x_ws, DMODEL, HIDN);
    }

    hipMemcpyAsync(d_out, x_ws, (size_t)MROWS * DMODEL * 4, hipMemcpyDeviceToDevice, stream);
}
